// SelfAttention_64072322121728
// MI455X (gfx1250) — matrix operand order repaired and checked by reading, not compiled, not run on hardware
//
#include <hip/hip_runtime.h>
#include <stdint.h>

typedef __attribute__((ext_vector_type(16))) __bf16 v16bf;
typedef __attribute__((ext_vector_type(8)))  __bf16 v8bf;
typedef __attribute__((ext_vector_type(4)))  __bf16 v4bf;
typedef __attribute__((ext_vector_type(8)))  float  v8f;
typedef __attribute__((__vector_size__(16))) int    v4i;

template<class X, class Y> struct SameT { static constexpr bool value = false; };
template<class X>          struct SameT<X, X> { static constexpr bool value = true; };

// CDNA5 async global->LDS copy (ASYNCcnt path), with safe fallback.
#if defined(__has_builtin)
#  if __has_builtin(__builtin_amdgcn_global_load_async_to_lds_b128)
#    define ATTN_HAVE_ASYNC 1
#  endif
#endif
#ifndef ATTN_HAVE_ASYNC
#  define ATTN_HAVE_ASYNC 0
#endif

#define LDSPAD 40  // 32 K-halves + 8 pad halves (80 B row stride, 16B-aligned)

// ---------------------------------------------------------------------------
// Generic NT/NN bf16 WMMA GEMM: C[M,N] = A[M,K] * op(B), fp32 accumulate.
//   A row-major K-fastest (float A converted to bf16 while staging).
//   B_NMAJOR == false: B is [N,K] row-major (NT).  true: B is [K,N] (NN),
//   transposed into LDS.  If gridDim.z==2, blockIdx.z==1 uses (B2, C2):
//   used to fuse the K and Q projections into one launch.
// 128x128 workgroup tile, 256 threads = 8 waves (2x4), 64x32 wave tile,
// K-chunk 32, double-buffered LDS, register- or async-staged global loads.
// ---------------------------------------------------------------------------
template<typename TA, typename TB, bool B_NMAJOR, typename TOUT>
__global__ __launch_bounds__(256)
void gemm_wmma(const TA* __restrict__ A, const TB* B, TOUT* C,
               const TB* B2, TOUT* C2,
               int M, int N, int K, int lda, int ldb, int ldc)
{
    if (blockIdx.z) { B = B2; C = C2; }

    __shared__ __bf16 ldsA[2][128][LDSPAD];
    __shared__ __bf16 ldsB[2][128][LDSPAD];

    constexpr bool asyncA = (ATTN_HAVE_ASYNC != 0) && !SameT<TA, float>::value;
    constexpr bool asyncB = (ATTN_HAVE_ASYNC != 0) && !B_NMAJOR && !SameT<TB, float>::value;

    const int  tid  = threadIdx.x;
    const int  lane = tid & 31;
    const int  wave = tid >> 5;
    const int  wm   = (wave >> 2) * 64;
    const int  wn   = (wave & 3) * 32;
    const int  lrow = lane & 15;
    const bool hi   = lane >= 16;

    const int m0 = blockIdx.y * 128;
    const int n0 = blockIdx.x * 128;

    const TA* Ab = A + (size_t)m0 * lda;
    const TB* Bb;
    if constexpr (B_NMAJOR) Bb = B + n0;                 // B[k][n], ldb = N
    else                    Bb = B + (size_t)n0 * ldb;   // B[n][k], ldb = K

    v8f acc[4][2] = {};

    // staging registers (unused flavors are optimized away)
    float4 aF[4]; v8bf aH[2];
    float4 bF[4]; v8bf bH[2]; float bS[16];

    auto issue_async = [&](const void* g, void* l) {
#if ATTN_HAVE_ASYNC
        __builtin_amdgcn_global_load_async_to_lds_b128(
            (__attribute__((address_space(1))) v4i*)g,
            (__attribute__((address_space(3))) v4i*)l, 0, 0);
#endif
    };

    // Phase 1: issue global loads (to regs, or async straight into LDS[buf]).
    auto load_tiles = [&](int k0, int buf) {
        if constexpr (SameT<TA, float>::value) {
            #pragma unroll
            for (int it = 0; it < 4; ++it) {
                int idx = tid + it * 256;
                aF[it] = *reinterpret_cast<const float4*>(
                    Ab + (size_t)(idx >> 3) * lda + k0 + ((idx & 7) << 2));
            }
        } else if constexpr (asyncA) {
            #pragma unroll
            for (int it = 0; it < 2; ++it) {
                int idx = tid + it * 256;
                int row = idx >> 2, ko = (idx & 3) << 3;
                issue_async(Ab + (size_t)row * lda + k0 + ko, &ldsA[buf][row][ko]);
            }
        } else {
            #pragma unroll
            for (int it = 0; it < 2; ++it) {
                int idx = tid + it * 256;
                aH[it] = *reinterpret_cast<const v8bf*>(
                    Ab + (size_t)(idx >> 2) * lda + k0 + ((idx & 3) << 3));
            }
        }
        if constexpr (B_NMAJOR) {
            #pragma unroll
            for (int it = 0; it < 16; ++it) {
                int idx = tid + it * 256;              // coalesced along n
                bS[it] = (float)Bb[(size_t)(k0 + (idx >> 7)) * ldb + (idx & 127)];
            }
        } else if constexpr (SameT<TB, float>::value) {
            #pragma unroll
            for (int it = 0; it < 4; ++it) {
                int idx = tid + it * 256;
                bF[it] = *reinterpret_cast<const float4*>(
                    Bb + (size_t)(idx >> 3) * ldb + k0 + ((idx & 7) << 2));
            }
        } else if constexpr (asyncB) {
            #pragma unroll
            for (int it = 0; it < 2; ++it) {
                int idx = tid + it * 256;
                int row = idx >> 2, ko = (idx & 3) << 3;
                issue_async(Bb + (size_t)row * ldb + k0 + ko, &ldsB[buf][row][ko]);
            }
        } else {
            #pragma unroll
            for (int it = 0; it < 2; ++it) {
                int idx = tid + it * 256;
                bH[it] = *reinterpret_cast<const v8bf*>(
                    Bb + (size_t)(idx >> 2) * ldb + k0 + ((idx & 3) << 3));
            }
        }
    };

    // Phase 2: convert + store regs into LDS[buf] (no-op for async operands).
    auto store_tiles = [&](int buf) {
        if constexpr (SameT<TA, float>::value) {
            #pragma unroll
            for (int it = 0; it < 4; ++it) {
                int idx = tid + it * 256;
                int row = idx >> 3, kq = (idx & 7) << 2;
                ldsA[buf][row][kq + 0] = (__bf16)aF[it].x;
                ldsA[buf][row][kq + 1] = (__bf16)aF[it].y;
                ldsA[buf][row][kq + 2] = (__bf16)aF[it].z;
                ldsA[buf][row][kq + 3] = (__bf16)aF[it].w;
            }
        } else if constexpr (!asyncA) {
            #pragma unroll
            for (int it = 0; it < 2; ++it) {
                int idx = tid + it * 256;
                *reinterpret_cast<v8bf*>(&ldsA[buf][idx >> 2][(idx & 3) << 3]) = aH[it];
            }
        }
        if constexpr (B_NMAJOR) {
            #pragma unroll
            for (int it = 0; it < 16; ++it) {
                int idx = tid + it * 256;
                ldsB[buf][idx & 127][idx >> 7] = (__bf16)bS[it];
            }
        } else if constexpr (SameT<TB, float>::value) {
            #pragma unroll
            for (int it = 0; it < 4; ++it) {
                int idx = tid + it * 256;
                int row = idx >> 3, kq = (idx & 7) << 2;
                ldsB[buf][row][kq + 0] = (__bf16)bF[it].x;
                ldsB[buf][row][kq + 1] = (__bf16)bF[it].y;
                ldsB[buf][row][kq + 2] = (__bf16)bF[it].z;
                ldsB[buf][row][kq + 3] = (__bf16)bF[it].w;
            }
        } else if constexpr (!asyncB) {
            #pragma unroll
            for (int it = 0; it < 2; ++it) {
                int idx = tid + it * 256;
                *reinterpret_cast<v8bf*>(&ldsB[buf][idx >> 2][(idx & 3) << 3]) = bH[it];
            }
        }
    };

    // Fragment loads per ISA 7.12.2 16-bit layouts + 8 WMMAs.
    auto compute = [&](int buf) {
        v16bf afrag[4], bfrag[2];
        const int aklo = hi ? 8 : 0;     // lanes>=16 hold K 8..15 / 24..31
        #pragma unroll
        for (int mi = 0; mi < 4; ++mi) {
            int r = wm + mi * 16 + lrow;
            union { v16bf v; v8bf h[2]; } u;
            u.h[0] = *reinterpret_cast<const v8bf*>(&ldsA[buf][r][aklo]);
            u.h[1] = *reinterpret_cast<const v8bf*>(&ldsA[buf][r][aklo + 16]);
            afrag[mi] = u.v;
        }
        const int bkb = hi ? 8 : 0;      // lanes>=16 hold K 8..15 / 24..31
        #pragma unroll
        for (int ni = 0; ni < 2; ++ni) {
            int c = wn + ni * 16 + lrow;
            union { v16bf v; v8bf h[2]; } u;
            u.h[0] = *reinterpret_cast<const v8bf*>(&ldsB[buf][c][bkb]);
            u.h[1] = *reinterpret_cast<const v8bf*>(&ldsB[buf][c][bkb + 16]);
            bfrag[ni] = u.v;
        }
        #pragma unroll
        for (int mi = 0; mi < 4; ++mi)
            #pragma unroll
            for (int ni = 0; ni < 2; ++ni)
                acc[mi][ni] = __builtin_amdgcn_wmma_f32_16x16x32_bf16(
                    false, afrag[mi], false, bfrag[ni],
                    (short)0, acc[mi][ni], false, false);
    };

    const int nk = K >> 5;

    // prologue: stage chunk 0 into buffer 0
    load_tiles(0, 0);
    store_tiles(0);

    for (int kc = 0; kc < nk; ++kc) {
        if constexpr (asyncA || asyncB)
            asm volatile("s_wait_asynccnt 0x0" ::: "memory");
        __syncthreads();

        const int cur = kc & 1;
        if (kc + 1 < nk) load_tiles((kc + 1) << 5, cur ^ 1);  // in flight during WMMA
        compute(cur);
        if (kc + 1 < nk) store_tiles(cur ^ 1);                // waits land here
    }

    // epilogue: C/D layout -> VGPR r holds (M=r | M=8+r), N = lane%16
    #pragma unroll
    for (int mi = 0; mi < 4; ++mi) {
        #pragma unroll
        for (int ni = 0; ni < 2; ++ni) {
            int col = n0 + wn + ni * 16 + lrow;
            #pragma unroll
            for (int r = 0; r < 8; ++r) {
                int row = m0 + wm + mi * 16 + (hi ? 8 + r : r);
                float v = acc[mi][ni][r];
                if constexpr (SameT<TOUT, float>::value)
                    C[(size_t)row * ldc + col] = v;
                else
                    C[(size_t)row * ldc + col] = (__bf16)v;
            }
        }
    }
}

// ---------------------------------------------------------------------------
// Column softmax stats (softmax over axis l per column m), two-phase for
// parallelism: 128-row partials (128 WGs) then online max/sum-exp merge.
// ---------------------------------------------------------------------------
__global__ __launch_bounds__(256)
void colstats_phase1(const float* __restrict__ S,
                     float* __restrict__ pmx, float* __restrict__ psm)
{
    const int L = 2048, RCHUNK = 128;
    int m  = blockIdx.x * 256 + threadIdx.x;
    int l0 = blockIdx.y * RCHUNK;
    float mx = -3.4e38f, s = 0.f;
    for (int l = l0; l < l0 + RCHUNK; ++l) {
        float v = S[(size_t)l * L + m];
        if (v > mx) { s = s * __expf(mx - v) + 1.f; mx = v; }
        else        { s += __expf(v - mx); }
    }
    pmx[(size_t)blockIdx.y * L + m] = mx;
    psm[(size_t)blockIdx.y * L + m] = s;
}

__global__ __launch_bounds__(256)
void colstats_phase2(const float* __restrict__ pmx, const float* __restrict__ psm,
                     float* __restrict__ cmax, float* __restrict__ rdenom)
{
    const int L = 2048, NCH = 16;
    int m = blockIdx.x * 256 + threadIdx.x;
    float mx = -3.4e38f, s = 0.f;
    for (int c = 0; c < NCH; ++c) {
        float m2 = pmx[(size_t)c * L + m];
        float s2 = psm[(size_t)c * L + m];
        if (m2 > mx) { s = s * __expf(mx - m2) + s2; mx = m2; }
        else         { s += s2 * __expf(m2 - mx); }
    }
    cmax[m]   = mx;
    rdenom[m] = 1.f / s;
}

// P[l,m] = bf16( exp(S[l,m]-cmax[m]) * rdenom[m] )  -- 1/D folded in
__global__ __launch_bounds__(256)
void expscale_kernel(const float* __restrict__ S, const float* __restrict__ cmax,
                     const float* __restrict__ rdenom, __bf16* __restrict__ P)
{
    size_t i4 = ((size_t)blockIdx.x * 256 + threadIdx.x) * 4;
    int m = (int)(i4 & 2047);
    const float4 s  = *reinterpret_cast<const float4*>(S + i4);
    const float4 cm = *reinterpret_cast<const float4*>(cmax + m);
    const float4 rd = *reinterpret_cast<const float4*>(rdenom + m);
    v4bf p;
    p[0] = (__bf16)(__expf(s.x - cm.x) * rd.x);
    p[1] = (__bf16)(__expf(s.y - cm.y) * rd.y);
    p[2] = (__bf16)(__expf(s.z - cm.z) * rd.z);
    p[3] = (__bf16)(__expf(s.w - cm.w) * rd.w);
    *reinterpret_cast<v4bf*>(P + i4) = p;
}

// ---------------------------------------------------------------------------
extern "C" void kernel_launch(void* const* d_in, const int* in_sizes, int n_in,
                              void* d_out, int out_size, void* d_ws, size_t ws_size,
                              hipStream_t stream)
{
    const int N = 16, L = 2048, C = 1024, A = 1024;
    const float* x  = (const float*)d_in[0];
    const float* Wq = (const float*)d_in[1];
    const float* Wk = (const float*)d_in[2];
    float* out = (float*)d_out;

    // workspace (per-batch buffers reused across batches; ~32.6 MB)
    char* ws = (char*)d_ws;
    __bf16* Kb  = (__bf16*)ws;  ws += (size_t)L * A * sizeof(__bf16);   // 4 MB
    __bf16* Qb  = (__bf16*)ws;  ws += (size_t)L * A * sizeof(__bf16);   // 4 MB
    float*  S   = (float*) ws;  ws += (size_t)L * L * sizeof(float);    // 16 MB
    __bf16* Pb  = (__bf16*)ws;  ws += (size_t)L * L * sizeof(__bf16);   // 8 MB
    float*  pmx = (float*) ws;  ws += (size_t)16 * L * sizeof(float);   // 128 KB
    float*  psm = (float*) ws;  ws += (size_t)16 * L * sizeof(float);   // 128 KB
    float*  cmx = (float*) ws;  ws += (size_t)L * sizeof(float);
    float*  rdn = (float*) ws;

    dim3 blk(256);
    for (int n = 0; n < N; ++n) {
        const float* xn = x   + (size_t)n * L * C;
        float*       on = out + (size_t)n * L * C;

        // K = x Wk^T and Q = x Wq^T fused via blockIdx.z (256 WGs)
        gemm_wmma<float, float, false, __bf16>
            <<<dim3(A / 128, L / 128, 2), blk, 0, stream>>>(
                xn, Wk, Kb, Wq, Qb, L, A, C, C, C, A);

        // S = K Q^T  (NT bf16 -> fp32)
        gemm_wmma<__bf16, __bf16, false, float>
            <<<dim3(L / 128, L / 128), blk, 0, stream>>>(
                Kb, Qb, S, Qb, S, L, L, A, A, A, L);

        // column-axis softmax stats, then P = exp(S - M) * (1/D) in bf16
        colstats_phase1<<<dim3(L / 256, 16), blk, 0, stream>>>(S, pmx, psm);
        colstats_phase2<<<dim3(L / 256), blk, 0, stream>>>(pmx, psm, cmx, rdn);
        expscale_kernel<<<dim3((L * L) / (256 * 4)), blk, 0, stream>>>(S, cmx, rdn, Pb);

        // out = P x  (NN: B = x fp32 [m,c], transposed into LDS)
        gemm_wmma<__bf16, float, true, float>
            <<<dim3(C / 128, L / 128), blk, 0, stream>>>(
                Pb, xn, on, xn, on, L, C, L, L, C, C);
    }
}
